// LSTMPredictor_75935021794085
// MI455X (gfx1250) — matrix operand order repaired and checked by reading, not compiled, not run on hardware
//
#include <hip/hip_runtime.h>

// ---------------------------------------------------------------------------
// 2-layer LSTM (B=256, H=512, T=1024) for MI455X / gfx1250.
//  - fp16 WMMA (v_wmma_f32_16x16x32_f16), fp32 accumulate, c-state in VGPRs
//  - 32 persistent WGs; each WG owns 16 hidden units; its 192KB weight slice
//    is DMA'd via the Tensor Data Mover into LDS (CDNA5 320KB) and stays there
//  - double-buffered h1/h2 state in workspace; one device-scope barrier/step
// ---------------------------------------------------------------------------

typedef __attribute__((ext_vector_type(16))) _Float16 v16h;
typedef __attribute__((ext_vector_type(8)))  float    v8f;
typedef __attribute__((ext_vector_type(4)))  unsigned u32x4;
typedef __attribute__((ext_vector_type(8)))  int      i32x8;
typedef __attribute__((ext_vector_type(4)))  int      i32x4;

#define BATCH 256
#define HID   512
#define SEQT  1024
#define NWG   32
#define HWG   16                    // hidden units per WG
#define KT    16                    // 512 / 32 k-tiles
#define WSLICE (4 * 16 * 32 * 16)   // halves per (matrix, WG) slice = 64KB
#define MATPK  (NWG * WSLICE)       // halves per packed matrix = 2MB

// LDS carve sizes (bytes)
#define SM_W      (WSLICE * 2)          // 65536 per matrix
#define SM_HSTG   (BATCH * HWG * 2)     // 8192
#define SM_XSTG   (BATCH * 4)           // 1024
#define SM_SMALL  (64*4 + 64*4 + 64*4 + HID*4 + 16)
#define SMEM_BYTES (3*SM_W + SM_HSTG + SM_XSTG + SM_SMALL)

union Frag { v16h v; u32x4 q[2]; };

__device__ __forceinline__ float sigmf(float v) { return 1.0f / (1.0f + __expf(-v)); }

// LDS byte offset of a generic pointer that points into shared memory
// (addrspace(3) ptrtoint gives the workgroup-relative LDS address the D# wants).
typedef __attribute__((address_space(3))) char lds_char;
__device__ __forceinline__ unsigned lds_offset_of(const void* p) {
  return (unsigned)(size_t)(lds_char*)p;
}

// A-fragment (16x32 f16) per ISA 7.12.2: lanes 0-15 -> M=lane, K = {0..7,16..23};
// lanes 16-31 -> M=lane-16, K = {8..15,24..31}.  Source: row-major h [256][512].
__device__ __forceinline__ v16h load_afrag(const _Float16* __restrict__ hb,
                                           int M, int kt, int lhalf) {
  Frag u;
  const _Float16* row = hb + ((size_t)M << 9) + (kt << 5);
  u.q[0] = *(const u32x4*)(row + (lhalf ? 8  : 0));
  u.q[1] = *(const u32x4*)(row + (lhalf ? 24 : 16));
  return u.v;
}

// B-fragment (32x16 f16): packed as [g][kt][lane][16] halves -> 32B/lane.
__device__ __forceinline__ v16h load_bfrag(const _Float16* w, int g, int kt, int lane) {
  Frag u;
  const _Float16* p = w + (size_t)((((g << 4) + kt) << 5) + lane) * 16;
  u.q[0] = *(const u32x4*)(p);
  u.q[1] = *(const u32x4*)(p + 8);
  return u.v;
}

#define WMMA_F16(A, B, C) \
  __builtin_amdgcn_wmma_f32_16x16x32_f16(false, (A), false, (B), (short)0, (C), false, false)

__device__ __forceinline__ void grid_barrier(unsigned* cnt, unsigned* gen) {
  __syncthreads();
  if (threadIdx.x == 0) {
    __threadfence();   // release this WG's h-state writes to device scope
    unsigned g = __hip_atomic_load(gen, __ATOMIC_RELAXED, __HIP_MEMORY_SCOPE_AGENT);
    unsigned a = __hip_atomic_fetch_add(cnt, 1u, __ATOMIC_ACQ_REL, __HIP_MEMORY_SCOPE_AGENT);
    if (a == (unsigned)(NWG - 1)) {
      __hip_atomic_store(cnt, 0u, __ATOMIC_RELAXED, __HIP_MEMORY_SCOPE_AGENT);
      __hip_atomic_store(gen, g + 1u, __ATOMIC_RELEASE, __HIP_MEMORY_SCOPE_AGENT);
    } else {
      while (__hip_atomic_load(gen, __ATOMIC_ACQUIRE, __HIP_MEMORY_SCOPE_AGENT) == g)
        __builtin_amdgcn_s_sleep(2);
    }
    __threadfence();   // acquire other WGs' writes
  }
  __syncthreads();
}

// Issue one TDM descriptor: contiguous 64KB (8192 x 8B) global -> LDS.
// D# per CDNA5 ISA ch.8: group0 = {flags, lds_addr, global_addr, type=2},
// group1 = {data_size=8B, tensor_dim0=8192, tile_dim0=8192, tile_dim1=1, stride}.
__device__ __forceinline__ void tdm_load_64k(const void* gsrc, unsigned lds_addr) {
#if __has_builtin(__builtin_amdgcn_tensor_load_to_lds)
  unsigned long long ga = (unsigned long long)gsrc;
  u32x4 g0 = { 0x1u,                                   // count=1 user descriptor
               lds_addr,                               // lds_addr (bytes)
               (unsigned)(ga & 0xffffffffu),           // global_addr[31:0]
               (unsigned)((ga >> 32) & 0x1ffffffu) | (2u << 30) }; // [56:32] | type=2
  i32x8 g1 = { (int)(3u << 16),          // workgroup_mask=0, data_size=3 (8B)
               (int)(8192u << 16),       // tensor_dim0[15:0] -> bits 63:48
               (int)(1u << 16),          // tensor_dim0 hi=0, tensor_dim1=1
               (int)(8192u << 16),       // tensor_dim1 hi=0, tile_dim0=8192
               1,                        // tile_dim1=1, tile_dim2=0 (2D tile)
               8192,                     // tensor_dim0_stride lo32
               0, 0 };                   // stride hi / dim1_stride = 0
  i32x4 gz = { 0, 0, 0, 0 };             // groups 2/3 unused (tile_dim2=0)
#if defined(__clang_major__) && (__clang_major__ >= 23)
  i32x8 gz8 = { 0, 0, 0, 0, 0, 0, 0, 0 };
  __builtin_amdgcn_tensor_load_to_lds(g0, g1, gz, gz, gz8, 0);
#else
  __builtin_amdgcn_tensor_load_to_lds(g0, g1, gz, gz, 0);
#endif
#endif
}

// Convert one fp32 weight matrix W[2048][512] (gates = h @ W^T, so B[k][n] = W[n][k])
// into packed fp16 B-fragments: dst[wg][g][kt][lane][e].
__global__ void pack_weights_kernel(const float* __restrict__ w, _Float16* __restrict__ dst) {
  unsigned id   = blockIdx.x * 256u + threadIdx.x;     // 2^20 elements
  int e    = id & 15;
  int lane = (id >> 4) & 31;
  int kt   = (id >> 9) & 15;
  int g    = (id >> 13) & 3;
  int wg   = (id >> 15) & 31;
  int n = g * HID + wg * HWG + (lane & 15);
  int k = kt * 32 + ((lane < 16) ? 0 : 8) + (e & 7) + ((e & 8) ? 16 : 0);
  dst[id] = (_Float16)w[(size_t)n * HID + k];
}

// Zero h1/h2 buffers (1MB contiguous) and reset the barrier.
__global__ void init_kernel(u32x4* __restrict__ hz, unsigned* __restrict__ bar) {
  unsigned i = blockIdx.x * 256u + threadIdx.x;        // 65536 x 16B = 1MB
  u32x4 z = {0u, 0u, 0u, 0u};
  hz[i] = z;
  if (i == 0) { bar[0] = 0u; bar[1] = 0u; }
}

__global__ __launch_bounds__(256, 1)
void lstm_persistent_kernel(const float* __restrict__ x,
                            const float* __restrict__ wih1,
                            const float* __restrict__ b1,
                            const float* __restrict__ b2,
                            const float* __restrict__ wlin,
                            const float* __restrict__ blin,
                            const _Float16* __restrict__ pk,   // [3][NWG][WSLICE]
                            _Float16* __restrict__ h1buf,      // [2][256][512]
                            _Float16* __restrict__ h2buf,      // [2][256][512]
                            unsigned* __restrict__ bar,
                            float* __restrict__ out)           // [256][1024]
{
  extern __shared__ char smem[];
  _Float16* w1     = (_Float16*)(smem);
  _Float16* wi2    = (_Float16*)(smem + SM_W);
  _Float16* wh2    = (_Float16*)(smem + 2*SM_W);
  _Float16* hstage = (_Float16*)(smem + 3*SM_W);
  float*    xstage = (float*)(smem + 3*SM_W + SM_HSTG);
  float*    b1s    = xstage + BATCH;
  float*    b2s    = b1s + 64;
  float*    wih1s  = b2s + 64;
  float*    wlins  = wih1s + 64;
  float*    blins  = wlins + HID;

  const int tid   = threadIdx.x;
  const int wg    = blockIdx.x;
  const int lane  = tid & 31;
  const int wv    = tid >> 5;
  const int lhalf = (lane >= 16) ? 1 : 0;
  const int c     = lane & 15;
  const int j0    = wg * HWG;
  const int rt0   = wv * 2, rt1 = wv * 2 + 1;

  // ---- one-time: park this WG's weight slices (3 x 64KB) in LDS ----
#if __has_builtin(__builtin_amdgcn_tensor_load_to_lds)
  if (wv == 0) {   // one wave issues the three TDM descriptors (EXEC-independent)
    _Float16* dsts[3] = { w1, wi2, wh2 };
    for (int m = 0; m < 3; ++m)
      tdm_load_64k(pk + (size_t)(m * NWG + wg) * WSLICE, lds_offset_of(dsts[m]));
    __builtin_amdgcn_s_wait_tensorcnt((short)0);
  }
#else
  {
    _Float16* dsts[3] = { w1, wi2, wh2 };
    for (int m = 0; m < 3; ++m) {
      const u32x4* s = (const u32x4*)(pk + (size_t)(m * NWG + wg) * WSLICE);
      u32x4* d = (u32x4*)dsts[m];
      for (int i = tid; i < WSLICE / 8; i += 256) d[i] = s[i];   // 16B chunks
    }
  }
#endif
  if (tid < 64) {
    int g = tid >> 4, cc = tid & 15;
    int ncol = g * HID + j0 + cc;
    b1s[tid]   = b1[ncol];
    b2s[tid]   = b2[ncol];
    wih1s[tid] = wih1[ncol];     // W_ih1 is [4H,1]
  }
  for (int i = tid; i < HID; i += 256) wlins[i] = wlin[i];
  if (tid == 0) blins[0] = blin[0];
  __syncthreads();

  v8f zerov = {0.f,0.f,0.f,0.f,0.f,0.f,0.f,0.f};
  v8f c1[2] = { zerov, zerov };
  v8f c2[2] = { zerov, zerov };
  v8f acc[2][4];

  for (int t = 0; t < SEQT; ++t) {
    const int p = t & 1;
    const _Float16* h1r = h1buf + (size_t)p       * BATCH * HID;   // h1_{t-1}
    _Float16*       h1w = h1buf + (size_t)(1 - p) * BATCH * HID;   // h1_t
    const _Float16* h2r = h2buf + (size_t)p       * BATCH * HID;   // h2_{t-1}
    _Float16*       h2w = h2buf + (size_t)(1 - p) * BATCH * HID;   // h2_t

    xstage[tid] = x[(size_t)tid * SEQT + t];
    if (t + 1 < SEQT) __builtin_prefetch(x + (size_t)tid * SEQT + t + 1, 0, 1);
    __syncthreads();

    // ---------- layer 1: gates1 = h1_{t-1} @ W_hh1^T (+ x*w_ih1 + b1) ----------
    for (int i = 0; i < 2; ++i)
      for (int g = 0; g < 4; ++g) acc[i][g] = zerov;
#pragma unroll 2
    for (int kt = 0; kt < KT; ++kt) {
      v16h a0 = load_afrag(h1r, rt0 * 16 + c, kt, lhalf);
      v16h a1 = load_afrag(h1r, rt1 * 16 + c, kt, lhalf);
#pragma unroll
      for (int g = 0; g < 4; ++g) {
        v16h b = load_bfrag(w1, g, kt, lane);
        acc[0][g] = WMMA_F16(a0, b, acc[0][g]);
        acc[1][g] = WMMA_F16(a1, b, acc[1][g]);
      }
    }
#pragma unroll
    for (int i = 0; i < 2; ++i) {
      const int rt = wv * 2 + i;
#pragma unroll
      for (int r = 0; r < 8; ++r) {
        const int M = rt * 16 + r + lhalf * 8;     // C/D layout: lane half -> M+8
        const float xv = xstage[M];
        float iv = sigmf(acc[i][0][r] + b1s[c]      + xv * wih1s[c]);
        float fv = sigmf(acc[i][1][r] + b1s[16 + c] + xv * wih1s[16 + c]);
        float gv = tanhf(acc[i][2][r] + b1s[32 + c] + xv * wih1s[32 + c]);
        float ov = sigmf(acc[i][3][r] + b1s[48 + c] + xv * wih1s[48 + c]);
        float cn = fv * c1[i][r] + iv * gv;
        c1[i][r] = cn;
        hstage[M * HWG + c] = (_Float16)(ov * tanhf(cn));
      }
    }
    __syncthreads();
    {   // publish h1_t slice: row tid, cols [j0, j0+16) -> 32B coalesced
      const u32x4* hs = (const u32x4*)(hstage + tid * HWG);
      u32x4* dst = (u32x4*)(h1w + ((size_t)tid << 9) + j0);
      dst[0] = hs[0]; dst[1] = hs[1];
    }

    grid_barrier(bar, bar + 1);   // h1_t now globally visible

    // WG0: emit out[:, t-1] from h2_{t-1} (visible after barrier); fixed fp order
    if (wg == 0 && t > 0) {
      const _Float16* hrow = h2r + ((size_t)tid << 9);
      float s = blins[0];
      for (int k = 0; k < HID; ++k) s += (float)hrow[k] * wlins[k];
      out[(size_t)tid * SEQT + (t - 1)] = s;
    }

    // ---------- layer 2: gates2 = h1_t @ W_ih2^T + h2_{t-1} @ W_hh2^T + b2 ----------
    for (int i = 0; i < 2; ++i)
      for (int g = 0; g < 4; ++g) acc[i][g] = zerov;
#pragma unroll 2
    for (int kt = 0; kt < KT; ++kt) {
      v16h a0 = load_afrag(h1w, rt0 * 16 + c, kt, lhalf);
      v16h a1 = load_afrag(h1w, rt1 * 16 + c, kt, lhalf);
#pragma unroll
      for (int g = 0; g < 4; ++g) {
        v16h b = load_bfrag(wi2, g, kt, lane);
        acc[0][g] = WMMA_F16(a0, b, acc[0][g]);
        acc[1][g] = WMMA_F16(a1, b, acc[1][g]);
      }
      v16h a2 = load_afrag(h2r, rt0 * 16 + c, kt, lhalf);
      v16h a3 = load_afrag(h2r, rt1 * 16 + c, kt, lhalf);
#pragma unroll
      for (int g = 0; g < 4; ++g) {
        v16h b = load_bfrag(wh2, g, kt, lane);
        acc[0][g] = WMMA_F16(a2, b, acc[0][g]);
        acc[1][g] = WMMA_F16(a3, b, acc[1][g]);
      }
    }
#pragma unroll
    for (int i = 0; i < 2; ++i) {
      const int rt = wv * 2 + i;
#pragma unroll
      for (int r = 0; r < 8; ++r) {
        const int M = rt * 16 + r + lhalf * 8;
        float iv = sigmf(acc[i][0][r] + b2s[c]);
        float fv = sigmf(acc[i][1][r] + b2s[16 + c]);
        float gv = tanhf(acc[i][2][r] + b2s[32 + c]);
        float ov = sigmf(acc[i][3][r] + b2s[48 + c]);
        float cn = fv * c2[i][r] + iv * gv;
        c2[i][r] = cn;
        hstage[M * HWG + c] = (_Float16)(ov * tanhf(cn));
      }
    }
    __syncthreads();
    {   // publish h2_t slice
      const u32x4* hs = (const u32x4*)(hstage + tid * HWG);
      u32x4* dst = (u32x4*)(h2w + ((size_t)tid << 9) + j0);
      dst[0] = hs[0]; dst[1] = hs[1];
    }
  }

  grid_barrier(bar, bar + 1);
  if (wg == 0) {   // final column: h2_{T-1} lives in buffer (SEQT & 1) == 0
    const _Float16* hrow = h2buf + (size_t)(SEQT & 1) * BATCH * HID + ((size_t)tid << 9);
    float s = blins[0];
    for (int k = 0; k < HID; ++k) s += (float)hrow[k] * wlins[k];
    out[(size_t)tid * SEQT + (SEQT - 1)] = s;
  }
}

extern "C" void kernel_launch(void* const* d_in, const int* in_sizes, int n_in,
                              void* d_out, int out_size, void* d_ws, size_t ws_size,
                              hipStream_t stream) {
  const float* x     = (const float*)d_in[0];
  const float* W_ih1 = (const float*)d_in[1];
  const float* W_hh1 = (const float*)d_in[2];
  const float* b1    = (const float*)d_in[3];
  const float* W_ih2 = (const float*)d_in[4];
  const float* W_hh2 = (const float*)d_in[5];
  const float* b2    = (const float*)d_in[6];
  const float* W_lin = (const float*)d_in[7];
  const float* b_lin = (const float*)d_in[8];
  float* out = (float*)d_out;

  // workspace layout: [barrier 1KB][h1 x2 512KB][h2 x2 512KB][packed weights 6MB]
  char* ws = (char*)d_ws;
  unsigned*  bar   = (unsigned*)ws;
  _Float16*  h1buf = (_Float16*)(ws + 1024);
  _Float16*  h2buf = (_Float16*)(ws + 1024 + 2 * BATCH * HID * 2);
  _Float16*  pk    = (_Float16*)(ws + 1024 + 4 * BATCH * HID * 2);

  // one-time (per call) weight pack: fp32 -> fp16 WMMA B-fragments
  pack_weights_kernel<<<MATPK / 256, 256, 0, stream>>>(W_hh1, pk);
  pack_weights_kernel<<<MATPK / 256, 256, 0, stream>>>(W_ih2, pk + (size_t)MATPK);
  pack_weights_kernel<<<MATPK / 256, 256, 0, stream>>>(W_hh2, pk + (size_t)2 * MATPK);

  // zero h-state (h1[0], h2[0] must be 0 at t=0; just clear all 1MB) + barrier
  init_kernel<<<(4 * BATCH * HID * 2) / 16 / 256, 256, 0, stream>>>((u32x4*)h1buf, bar);

  lstm_persistent_kernel<<<NWG, 256, SMEM_BYTES, stream>>>(
      x, W_ih1, b1, b2, W_lin, b_lin, pk, h1buf, h2buf, bar, out);
}
